// LightweightCrossViewFusion_56169582297652
// MI455X (gfx1250) — matrix operand order repaired and checked by reading, not compiled, not run on hardware
//
#include <hip/hip_runtime.h>
#include <hip/hip_bf16.h>

#define FEAT   256
#define LTOK   16384
#define NVIEW  8
#define GHID   64
#define TL     2      // token positions per block
#define MTILE  16     // rows per block (TL * NVIEW)

typedef __bf16 bf16;
typedef __attribute__((ext_vector_type(16))) __bf16 v16bf;
typedef __attribute__((ext_vector_type(8)))  float  v8f;

__device__ __forceinline__ v8f vzero8() {
  v8f z;
#pragma unroll
  for (int e = 0; e < 8; ++e) z[e] = 0.0f;
  return z;
}

// A fragment (16x32 bf16, M x K) gathered from an LDS tile, row-major, leading dim ldk.
// Layout: lane m(0..15): K 0..15 contiguous ; lanes 16..31: K 16..31 (matches B fragment).
__device__ __forceinline__ v16bf lds_a_frag(const bf16* base, int lane, int kstep, int ldk) {
  const int row  = lane & 15;
  const int half = lane >> 4;
  const bf16* p0 = base + row * ldk + kstep * 32 + half * 16;
  const bf16* p1 = p0 + 8;
  v16bf a;
#pragma unroll
  for (int e = 0; e < 8; ++e) { a[e] = p0[e]; a[e + 8] = p1[e]; }
  return a;
}

// B fragment (32x16 bf16, K x N) where B = W^T, W row-major [Nrows][ldk] in global (L2-resident).
// Layout used here (same as A): lanes 0..15 hold K 0..15 of column (lane), lanes 16..31 hold K 16..31.
// => each lane reads 16 contiguous bf16 (32B) of one weight row.
__device__ __forceinline__ v16bf glb_b_frag(const bf16* __restrict__ W, int lane, int jt,
                                            int kstep, int ldk) {
  const int col = jt * 16 + (lane & 15);
  const int k0  = kstep * 32 + (lane >> 4) * 16;
  return *(const v16bf*)(W + col * ldk + k0);
}

__device__ __forceinline__ v8f wmma_bf16(v16bf a, v16bf b, v8f c) {
  return __builtin_amdgcn_wmma_f32_16x16x32_bf16(false, a, false, b, (short)0, c, false, false);
}

// ---------------------------------------------------------------------------
// Setup: weights -> bf16 in workspace; view_info[n][c] = (n/8)*rowsum(Wv)[c] + bv[c]
// ---------------------------------------------------------------------------
__global__ __launch_bounds__(256)
void setup_kernel(const float* __restrict__ vw, const float* __restrict__ vb,
                  const float* __restrict__ inw, const float* __restrict__ outw,
                  const float* __restrict__ g1w,
                  bf16* __restrict__ inwbf, bf16* __restrict__ outwbf,
                  bf16* __restrict__ g1wbf, float* __restrict__ vinfo) {
  const int tid    = blockIdx.x * blockDim.x + threadIdx.x;
  const int stride = gridDim.x * blockDim.x;
  for (int i = tid; i < 3 * FEAT * FEAT; i += stride) inwbf[i]  = (bf16)inw[i];
  for (int i = tid; i < FEAT * FEAT;     i += stride) outwbf[i] = (bf16)outw[i];
  for (int i = tid; i < GHID * FEAT;     i += stride) g1wbf[i]  = (bf16)g1w[i];
  if (tid < FEAT) {
    float s = 0.0f;
    for (int k = 0; k < FEAT; ++k) s += vw[tid * FEAT + k];
    for (int n = 0; n < NVIEW; ++n)
      vinfo[n * FEAT + tid] = (float)n * 0.125f * s + vb[tid];
  }
}

// ---------------------------------------------------------------------------
// Fused pipeline: one block = 2 token positions (16 rows), 8 waves of 32.
// ---------------------------------------------------------------------------
__global__ __launch_bounds__(256)
void fusion_kernel(const float* __restrict__ feat,   const float* __restrict__ vinfo,
                   const bf16*  __restrict__ Wqkv,   const float* __restrict__ Bqkv,
                   const bf16*  __restrict__ Wout,   const float* __restrict__ Bout,
                   const bf16*  __restrict__ Wg1,    const float* __restrict__ Bg1,
                   const float* __restrict__ Wg2,    const float* __restrict__ Bg2,
                   const float* __restrict__ lng,    const float* __restrict__ lnb,
                   float* __restrict__ out) {
  __shared__ __align__(16) unsigned char smem[59648];
  bf16*  Xs    = (bf16*)smem;                        // [16][256] bf16 (later: ctx bf16)
  float* qkv   = (float*)(smem + 8192);              // [16][768] f32
  float* attf  = (float*)(smem + 8192);              // overlay after attention: [16][256]
  bf16*  attb  = (bf16*)(smem + 8192 + 16384);       // [16][256] bf16
  float* g1s   = (float*)(smem + 8192 + 24576);      // [16][64]
  float* fus   = (float*)(smem + 8192 + 28672);      // [2][256]
  float* attnS = (float*)(smem + 57344);             // [8 waves][64]
  float* g2s   = (float*)(smem + 59392);             // [16]
  float* wls   = g2s + 16;                           // [16]
  float* stat  = wls + 16;                           // [4]: mu/rsig per l

  const int tid  = threadIdx.x;
  const int lane = tid & 31;
  const int wave = tid >> 5;
  const int l0   = blockIdx.x * TL;

  // ---- Stage X = features + view_info, converted to bf16 ----
  {
    const int r  = tid >> 4;            // 0..15
    const int kb = (tid & 15) * 16;
    const int li = r >> 3;
    const int n  = r & 7;
    const float* fp = feat + (size_t)n * LTOK * FEAT + (size_t)(l0 + li) * FEAT;
    const float* vp = vinfo + n * FEAT;
#pragma unroll
    for (int e = 0; e < 16; ++e) {
      const int k = kb + e;
      Xs[r * FEAT + k] = (bf16)(fp[k] + vp[k]);
    }
  }
  __syncthreads();

  // ---- GEMM1: qkv[16][768] = X @ Wqkv^T + b  (wave owns 6 N-tiles) ----
  {
    v8f acc[6];
#pragma unroll
    for (int t = 0; t < 6; ++t) acc[t] = vzero8();
#pragma unroll
    for (int ks = 0; ks < 8; ++ks) {
      const v16bf a = lds_a_frag(Xs, lane, ks, FEAT);
#pragma unroll
      for (int t = 0; t < 6; ++t) {
        const v16bf b = glb_b_frag(Wqkv, lane, wave * 6 + t, ks, FEAT);
        acc[t] = wmma_bf16(a, b, acc[t]);
      }
    }
    const int row0 = (lane >> 4) * 8;
#pragma unroll
    for (int t = 0; t < 6; ++t) {
      const int col = (wave * 6 + t) * 16 + (lane & 15);
      const float bia = Bqkv[col];
#pragma unroll
      for (int r = 0; r < 8; ++r)
        qkv[(row0 + r) * 768 + col] = acc[t][r] + bia;
    }
  }
  __syncthreads();

  // ---- Attention: wave -> (l = wave>>2, head = wave&3); seq = 8 views ----
  {
    const int li = wave >> 2, h = wave & 3;
    const float* qb = qkv + (li * 8) * 768 + h * 64;
    const float* kb = qkv + (li * 8) * 768 + 256 + h * 64;
    const int n0 = lane >> 3, m0 = lane & 7, n1 = n0 + 4;
    float s0 = 0.0f, s1 = 0.0f;
#pragma unroll 8
    for (int d = 0; d < 64; ++d) {
      const float kv = kb[m0 * 768 + d];
      s0 += qb[n0 * 768 + d] * kv;
      s1 += qb[n1 * 768 + d] * kv;
    }
    s0 *= 0.125f; s1 *= 0.125f;   // 1/sqrt(64)
    float mx0 = s0, mx1 = s1;
#pragma unroll
    for (int o = 1; o < 8; o <<= 1) {
      mx0 = fmaxf(mx0, __shfl_xor(mx0, o, 32));
      mx1 = fmaxf(mx1, __shfl_xor(mx1, o, 32));
    }
    const float e0 = __expf(s0 - mx0), e1 = __expf(s1 - mx1);
    float su0 = e0, su1 = e1;
#pragma unroll
    for (int o = 1; o < 8; o <<= 1) {
      su0 += __shfl_xor(su0, o, 32);
      su1 += __shfl_xor(su1, o, 32);
    }
    attnS[wave * 64 + n0 * 8 + m0] = e0 / su0;
    attnS[wave * 64 + n1 * 8 + m0] = e1 / su1;
  }
  __syncthreads();
  {
    const int li = wave >> 2, h = wave & 3;
    const float* vbp = qkv + (li * 8) * 768 + 512 + h * 64;
    bf16* ctx = Xs;   // reuse Xs region as bf16 ctx tile [16][256]
#pragma unroll
    for (int dd = 0; dd < 2; ++dd) {
      const int d = lane + dd * 32;
#pragma unroll
      for (int n = 0; n < 8; ++n) {
        float s = 0.0f;
#pragma unroll
        for (int m = 0; m < 8; ++m) s += attnS[wave * 64 + n * 8 + m] * vbp[m * 768 + d];
        ctx[(li * 8 + n) * FEAT + h * 64 + d] = (bf16)s;
      }
    }
  }
  __syncthreads();

  // ---- GEMM2: attended[16][256] = ctx @ Wout^T + b  (wave owns 2 N-tiles) ----
  {
    v8f acc[2];
    acc[0] = vzero8(); acc[1] = vzero8();
#pragma unroll
    for (int ks = 0; ks < 8; ++ks) {
      const v16bf a = lds_a_frag(Xs, lane, ks, FEAT);
#pragma unroll
      for (int t = 0; t < 2; ++t) {
        const v16bf b = glb_b_frag(Wout, lane, wave * 2 + t, ks, FEAT);
        acc[t] = wmma_bf16(a, b, acc[t]);
      }
    }
    const int row0 = (lane >> 4) * 8;
#pragma unroll
    for (int t = 0; t < 2; ++t) {
      const int col = (wave * 2 + t) * 16 + (lane & 15);
      const float bia = Bout[col];
#pragma unroll
      for (int r = 0; r < 8; ++r) {
        const float v = acc[t][r] + bia;
        attf[(row0 + r) * FEAT + col] = v;        // f32 copy (gate aggregation + fuse)
        attb[(row0 + r) * FEAT + col] = (bf16)v;  // bf16 copy (gate GEMM A)
      }
    }
  }
  __syncthreads();

  // ---- Gate GEMM: g1[16][64] = relu(attended @ Wg1^T + b1), waves 0..3 ----
  if (wave < 4) {
    v8f acc = vzero8();
#pragma unroll
    for (int ks = 0; ks < 8; ++ks) {
      const v16bf a = lds_a_frag(attb, lane, ks, FEAT);
      const v16bf b = glb_b_frag(Wg1, lane, wave, ks, FEAT);
      acc = wmma_bf16(a, b, acc);
    }
    const int row0 = (lane >> 4) * 8;
    const int col  = wave * 16 + (lane & 15);
    const float bia = Bg1[col];
#pragma unroll
    for (int r = 0; r < 8; ++r)
      g1s[(row0 + r) * GHID + col] = fmaxf(acc[r] + bia, 0.0f);
  }
  __syncthreads();

  // ---- g2 = sigmoid(g1 @ w2 + b2) : 16 rows, wave 0 ----
  if (wave == 0) {
    const int row = lane & 15;
    float s = 0.0f;
#pragma unroll 8
    for (int j = 0; j < GHID; ++j) s += g1s[row * GHID + j] * Wg2[j];
    const float g = 1.0f / (1.0f + __expf(-(s + Bg2[0])));
    if (lane < 16) g2s[row] = g;
  }
  __syncthreads();

  // ---- softmax over the 8 views per token position, wave 0 ----
  if (wave == 0) {
    const int idx = lane & 15;          // l = idx>>3, n = idx&7
    const float g = g2s[idx];
    float mx = g;
#pragma unroll
    for (int o = 1; o < 8; o <<= 1) mx = fmaxf(mx, __shfl_xor(mx, o, 32));
    const float e = __expf(g - mx);
    float su = e;
#pragma unroll
    for (int o = 1; o < 8; o <<= 1) su += __shfl_xor(su, o, 32);
    if (lane < 16) wls[idx] = e / su;
  }
  __syncthreads();

  // ---- fused[l][c] = sum_n attended * w ----
  {
    const int li = tid >> 7;
    const int cb = tid & 127;
#pragma unroll
    for (int cc = 0; cc < 2; ++cc) {
      const int c = cb + cc * 128;
      float s = 0.0f;
#pragma unroll
      for (int n = 0; n < 8; ++n)
        s += attf[(li * 8 + n) * FEAT + c] * wls[li * 8 + n];
      fus[li * FEAT + c] = s;
    }
  }
  __syncthreads();

  // ---- LayerNorm stats: wave 0 -> l 0, wave 1 -> l 1 ----
  if (wave < 2) {
    float s = 0.0f, ss = 0.0f;
#pragma unroll
    for (int e = 0; e < 8; ++e) {
      const float v = fus[wave * FEAT + lane * 8 + e];
      s += v; ss += v * v;
    }
#pragma unroll
    for (int o = 1; o < 32; o <<= 1) { s += __shfl_xor(s, o, 32); ss += __shfl_xor(ss, o, 32); }
    if (lane == 0) {
      const float mu  = s * (1.0f / FEAT);
      const float var = ss * (1.0f / FEAT) - mu * mu;
      stat[wave * 2 + 0] = mu;
      stat[wave * 2 + 1] = rsqrtf(var + 1e-5f);
    }
  }
  __syncthreads();

  // ---- normalize + affine + store ----
  {
    const int li = tid >> 7;
    const int cb = tid & 127;
    const float mu = stat[li * 2 + 0], rs = stat[li * 2 + 1];
    float* op = out + (size_t)(l0 + li) * FEAT;
#pragma unroll
    for (int cc = 0; cc < 2; ++cc) {
      const int c = cb + cc * 128;
      op[c] = (fus[li * FEAT + c] - mu) * rs * lng[c] + lnb[c];
    }
  }
}

// ---------------------------------------------------------------------------
extern "C" void kernel_launch(void* const* d_in, const int* in_sizes, int n_in,
                              void* d_out, int out_size, void* d_ws, size_t ws_size,
                              hipStream_t stream) {
  (void)in_sizes; (void)n_in; (void)out_size; (void)ws_size;
  const float* feat = (const float*)d_in[0];
  const float* vw   = (const float*)d_in[1];
  const float* vb   = (const float*)d_in[2];
  const float* inw  = (const float*)d_in[3];
  const float* inb  = (const float*)d_in[4];
  const float* outw = (const float*)d_in[5];
  const float* outb = (const float*)d_in[6];
  const float* g1w  = (const float*)d_in[7];
  const float* g1b  = (const float*)d_in[8];
  const float* g2w  = (const float*)d_in[9];
  const float* g2b  = (const float*)d_in[10];
  const float* lng  = (const float*)d_in[11];
  const float* lnb  = (const float*)d_in[12];
  float* out = (float*)d_out;

  unsigned char* ws = (unsigned char*)d_ws;
  bf16*  inwbf  = (bf16*)(ws);                              // 3*256*256*2 = 393216 B
  bf16*  outwbf = (bf16*)(ws + 393216);                     // 256*256*2   = 131072 B
  bf16*  g1wbf  = (bf16*)(ws + 393216 + 131072);            // 64*256*2    =  32768 B
  float* vinfo  = (float*)(ws + 393216 + 131072 + 32768);   // 8*256*4     =   8192 B

  setup_kernel<<<256, 256, 0, stream>>>(vw, vb, inw, outw, g1w, inwbf, outwbf, g1wbf, vinfo);
  fusion_kernel<<<LTOK / TL, 256, 0, stream>>>(feat, vinfo,
                                               inwbf, inb, outwbf, outb,
                                               g1wbf, g1b, g2w, g2b,
                                               lng, lnb, out);
}
